// Decoder_79869211836601
// MI455X (gfx1250) — matrix operand order repaired and checked by reading, not compiled, not run on hardware
//
#include <hip/hip_runtime.h>
#include <hip/hip_bf16.h>

// ---------------------------------------------------------------------------
// CDNA5 (gfx1250) fused D-VAE decoder, round 2.
// - one persistent workgroup per 16 batch rows (64 blocks x 512 threads)
// - all GEMMs via v_wmma_f32_16x16x32_bf16, fragment-major packed bf16 weights
// - branch-free epilogues via zero-padded bias/gather tables (prep kernels)
// - hidden-state staging via GLOBAL_LOAD_ASYNC_TO_LDS_B128 + s_wait_asynccnt
// - global_prefetch_b8 on upcoming weight panels
// ---------------------------------------------------------------------------

typedef __attribute__((ext_vector_type(16))) __bf16 v16bf;
typedef __attribute__((ext_vector_type(8)))  __bf16 v8bf;
typedef __attribute__((ext_vector_type(8)))  float  v8f;

#define HDIM 501
#define H2T  1002
#define H4T  2004
#define LDIM 56
#define CDIM 8
#define SDIM 8
#define MTILE 16
#define NTHREADS 512
#define NWAVES 16
#define GH_STRIDE 1536
#define GEN_ENC_BASE 65536   // B*S*S

// ---- workspace layout (bf16 elements) ----
static constexpr size_t SZ_LIN1 = (size_t)64 * 512;
static constexpr size_t SZ_WHH  = (size_t)512 * 1536;   // N = 3 x 512 (segmented)
static constexpr size_t SZ_SQ   = (size_t)512 * 512;
static constexpr size_t SZ_AV1  = (size_t)512 * 1024;
static constexpr size_t SZ_AV2  = (size_t)1024 * 16;
static constexpr size_t SZ_AE1  = (size_t)1024 * 2016;  // K = 2 x 512 (segmented)
static constexpr size_t OFF_LIN1 = 0;
static constexpr size_t OFF_WHH  = OFF_LIN1 + SZ_LIN1;
static constexpr size_t OFF_GATE = OFF_WHH + SZ_WHH;
static constexpr size_t OFF_MAP  = OFF_GATE + SZ_SQ;
static constexpr size_t OFF_AV1  = OFF_MAP + SZ_SQ;
static constexpr size_t OFF_AV2  = OFF_AV1 + SZ_AV1;
static constexpr size_t OFF_AE1  = OFF_AV2 + SZ_AV2;
static constexpr size_t SZ_NHS   = (size_t)64 * SDIM * MTILE * 512;
static constexpr size_t OFF_NHS  = OFF_AE1 + SZ_AE1;
static constexpr size_t OFF_GATED= OFF_NHS + SZ_NHS;
static constexpr size_t BF16_TOTAL = OFF_GATED + SZ_NHS;
static constexpr size_t GH_BYTE_OFF = ((BF16_TOTAL * 2 + 255) / 256) * 256;
static constexpr size_t GH_F32 = (size_t)64 * MTILE * GH_STRIDE;
static constexpr size_t BIAS_BYTE_OFF = GH_BYTE_OFF + GH_F32 * 4;
// padded f32 tables (elements, relative to BIAS_BYTE_OFF)
static constexpr size_t POFF_LIN1B = 0;        // 512
static constexpr size_t POFF_AVB1  = 512;      // 1024
static constexpr size_t POFF_GATEB = 1536;     // 512
static constexpr size_t POFF_MAPB  = 2048;     // 512
static constexpr size_t POFF_AEB1  = 2560;     // 2016
static constexpr size_t POFF_AEW2  = 4576;     // 2016
static constexpr size_t POFF_BIH   = 6592;     // 1536
static constexpr size_t POFF_BHH   = 8128;     // 1536
static constexpr size_t POFF_WIH   = 9664;     // 1536*8 = 12288

// ---------------------------------------------------------------------------
__global__ void zero_f32_kernel(float* __restrict__ p, int n) {
    for (int i = blockIdx.x * blockDim.x + threadIdx.x; i < n; i += gridDim.x * blockDim.x)
        p[i] = 0.0f;
}

// dst[i] = i < n_orig ? src[i] : 0   (i < n_pad)
__global__ void pad_f32_kernel(const float* __restrict__ src, float* __restrict__ dst,
                               int n_orig, int n_pad) {
    for (int i = blockIdx.x * blockDim.x + threadIdx.x; i < n_pad; i += gridDim.x * blockDim.x)
        dst[i] = (i < n_orig) ? src[i] : 0.0f;
}

// pad 3 gate segments of 501 rows to 512 rows each; width = trailing dim (1 or 8)
__global__ void pad3_kernel(const float* __restrict__ src, float* __restrict__ dst,
                            int width) {
    int total = 3 * 512 * width;
    for (int i = blockIdx.x * blockDim.x + threadIdx.x; i < total; i += gridDim.x * blockDim.x) {
        int g = i / (512 * width);
        int r = i % (512 * width);
        int n = r / width, c = r % width;
        dst[i] = (n < HDIM) ? src[((size_t)g * HDIM + n) * width + c] : 0.0f;
    }
}

// Pack W into WMMA-B fragment-major bf16 with segmented padding on both dims.
// logical k: segment kg = k/KsegPad, kk = k%KsegPad; valid iff kk < KsegSrc.
// source index = (ng*NsegSrc+nn)*rowstride + (kg*KsegSrc+kk).
__global__ void pack_b_seg_kernel(const float* __restrict__ W, __bf16* __restrict__ out,
                                  int KsegSrc, int KsegPad, int NsegSrc, int NsegPad,
                                  int Kpad, int Npad, int rowstride) {
    const int nkt = Kpad >> 5;
    const long total = (long)nkt * (Npad >> 4) * 512;
    for (long t = blockIdx.x * (long)blockDim.x + threadIdx.x; t < total;
         t += (long)gridDim.x * blockDim.x) {
        long f = t >> 9; int w = (int)(t & 511);
        int lane = w >> 4, h = w & 15;
        int kt = (int)(f % nkt), nt = (int)(f / nkt);
        int k = kt * 32 + ((lane < 16) ? 0 : 8) + ((h < 8) ? h : 8 + h);
        int n = nt * 16 + (lane & 15);
        int kg = k / KsegPad, kk = k - kg * KsegPad;
        int ng = n / NsegPad, nn = n - ng * NsegPad;
        float v = 0.0f;
        if (kk < KsegSrc && nn < NsegSrc)
            v = W[((long)ng * NsegSrc + nn) * rowstride + (long)kg * KsegSrc + kk];
        out[t] = (__bf16)v;
    }
}

// ---------------------------------------------------------------------------
__device__ __forceinline__ v8f wmma_bf16(v16bf a, v16bf b, v8f c) {
    return __builtin_amdgcn_wmma_f32_16x16x32_bf16(false, a, false, b, (short)0, c,
                                                   false, false);
}

__device__ __forceinline__ v16bf load_a_frag(const __bf16* A, int astride, int kt, int lane) {
    int m = lane & 15;
    int klo = (lane < 16) ? 0 : 8;
    const __bf16* p = A + m * astride + kt * 32 + klo;
    v8bf lo = *(const v8bf*)(p);
    v8bf hi = *(const v8bf*)(p + 16);
    v16bf a;
#pragma unroll
    for (int i = 0; i < 8; ++i) { a[i] = lo[i]; a[i + 8] = hi[i]; }
    return a;
}

__device__ __forceinline__ v16bf load_b_frag(const __bf16* Bp, int nkt, int nt, int kt, int lane) {
    return *(const v16bf*)(Bp + (((long)nt * nkt + kt) * 32 + lane) * 16);
}

__device__ __forceinline__ float sigf(float x) { return 1.0f / (1.0f + __expf(-x)); }

// GRU 3H pre-activation GEMM: A = xbuf[0:512) bf16, B = Whh packed (N = 3x512)
__device__ __forceinline__ void gemm_gh(int wv, int lane, const __bf16* xbuf,
                                        const __bf16* whhp, float* ghws) {
    for (int nt = wv; nt < 96; nt += NWAVES) {
        if (nt + NWAVES < 96)
            __builtin_prefetch(whhp + ((long)(nt + NWAVES) * 16) * 512 + lane * 16, 0, 1);
        v8f c = {};
#pragma unroll 4
        for (int kt = 0; kt < 16; ++kt)
            c = wmma_bf16(load_a_frag(xbuf, 1024, kt, lane),
                          load_b_frag(whhp, 16, nt, kt, lane), c);
        int n = nt * 16 + (lane & 15);
        int mb = (lane >> 4) * 8;
#pragma unroll
        for (int r = 0; r < 8; ++r) ghws[(size_t)(mb + r) * GH_STRIDE + n] = c[r];
    }
}

// branch-free GRU combine; all tables zero-padded to 512-aligned segments
__device__ __forceinline__ void gru_combine(int tid,
                                            const float* wihp, const float* bihp,
                                            const float* bhhp, const float* ghws,
                                            bool use_gh, int hp_mode, float* hvf,
                                            const float* acc, __bf16* xbuf,
                                            const int* cids) {
    for (int t = tid; t < MTILE * 512; t += NTHREADS) {
        int m = t >> 9, n = t & 511;
        int c = cids[m];
        float ir = wihp[(size_t)n * CDIM + c] + bihp[n];
        float iz = wihp[(size_t)(512 + n) * CDIM + c] + bihp[512 + n];
        float in_ = wihp[(size_t)(1024 + n) * CDIM + c] + bihp[1024 + n];
        float hr = bhhp[n], hz = bhhp[512 + n], hn = bhhp[1024 + n];
        if (use_gh) {
            const float* g = ghws + (size_t)m * GH_STRIDE;
            hr += g[n]; hz += g[512 + n]; hn += g[1024 + n];
        }
        float r = sigf(ir + hr);
        float u = sigf(iz + hz);
        float nn = tanhf(in_ + r * hn);
        float hp = (hp_mode == 0) ? hvf[t] : ((hp_mode == 2) ? acc[t] : 0.0f);
        float v = (1.0f - u) * nn + u * hp;
        v = (n < HDIM) ? v : 0.0f;   // keep padded lanes exactly zero
        hvf[t] = v;
        xbuf[m * 1024 + n] = (__bf16)v;
    }
}

// ---------------------------------------------------------------------------
__global__ void __launch_bounds__(NTHREADS)
decoder_kernel(const float* __restrict__ z, const float* __restrict__ dep,
               const float* __restrict__ ne, const float* __restrict__ av_b2,
               const float* __restrict__ ae_b2, const float* __restrict__ ptab,
               const __bf16* __restrict__ lin1p, const __bf16* __restrict__ whhp,
               const __bf16* __restrict__ gatep, const __bf16* __restrict__ mapp,
               const __bf16* __restrict__ av1p, const __bf16* __restrict__ av2p,
               const __bf16* __restrict__ ae1p, __bf16* __restrict__ nhsws,
               __bf16* __restrict__ gatedws, float* __restrict__ ghall,
               float* __restrict__ out) {
    __shared__ __align__(32) __bf16 xbuf[MTILE * 1024];                  // A staging
    __shared__ __align__(32) unsigned char scratch_raw[MTILE * 512 * 4]; // acc | abuf
    __shared__ float hvf[MTILE * 512];
    __shared__ float ebuf[MTILE];
    __shared__ float depv[MTILE];
    __shared__ int cids[MTILE];

    float* acc = (float*)scratch_raw;
    __bf16* abuf = (__bf16*)scratch_raw;

    const float* lin1_bp = ptab + POFF_LIN1B;
    const float* av_b1p  = ptab + POFF_AVB1;
    const float* gate_bp = ptab + POFF_GATEB;
    const float* map_bp  = ptab + POFF_MAPB;
    const float* ae_b1p  = ptab + POFF_AEB1;
    const float* ae_w2p  = ptab + POFF_AEW2;
    const float* bihp    = ptab + POFF_BIH;
    const float* bhhp    = ptab + POFF_BHH;
    const float* wihp    = ptab + POFF_WIH;

    const int tid = threadIdx.x;
    const int lane = tid & 31;
    const int wv = tid >> 5;
    const int b0 = blockIdx.x * MTILE;
    float* ghws = ghall + (size_t)blockIdx.x * MTILE * GH_STRIDE;
    __bf16* nhsb = nhsws + (size_t)blockIdx.x * SDIM * MTILE * 512;
    __bf16* gatedb = gatedws + (size_t)blockIdx.x * SDIM * MTILE * 512;
    // low 32 bits of a generic LDS pointer are the LDS byte offset (flat aperture)
    const unsigned lds_xbuf = (unsigned)(unsigned long long)(uintptr_t)&xbuf[0];

    for (int t = tid; t < MTILE * 1024; t += NTHREADS) xbuf[t] = (__bf16)0.0f;
    __syncthreads();
    for (int t = tid; t < MTILE * LDIM; t += NTHREADS) {
        int m = t / LDIM, k = t % LDIM;
        xbuf[m * 1024 + k] = (__bf16)z[(size_t)(b0 + m) * LDIM + k];
    }
    __syncthreads();

    // ---- graph_state = z @ lin1_W^T + b : K=64 (2 kt), N=512 (32 nt) ----
    {
        v8f c0 = {}, c1 = {};
        int nt0 = wv, nt1 = wv + NWAVES;
#pragma unroll
        for (int kt = 0; kt < 2; ++kt) {
            v16bf a = load_a_frag(xbuf, 1024, kt, lane);
            c0 = wmma_bf16(a, load_b_frag(lin1p, 2, nt0, kt, lane), c0);
            c1 = wmma_bf16(a, load_b_frag(lin1p, 2, nt1, kt, lane), c1);
        }
        __syncthreads();   // output aliases A buffer
        int mb = (lane >> 4) * 8;
        int n0 = nt0 * 16 + (lane & 15), n1 = nt1 * 16 + (lane & 15);
        float bb0 = lin1_bp[n0], bb1 = lin1_bp[n1];
#pragma unroll
        for (int r = 0; r < 8; ++r) {
            float v0 = (n0 < HDIM) ? (c0[r] + bb0) : 0.0f;
            float v1 = (n1 < HDIM) ? (c1[r] + bb1) : 0.0f;
            hvf[(mb + r) * 512 + n0] = v0;  xbuf[(mb + r) * 1024 + n0] = (__bf16)v0;
            hvf[(mb + r) * 512 + n1] = v1;  xbuf[(mb + r) * 1024 + n1] = (__bf16)v1;
        }
        __syncthreads();
    }

    for (int index = 0; index < SDIM; ++index) {
        if (tid < MTILE) {
            const float* row = ne + (size_t)(b0 + tid) * (SDIM * CDIM) + index * CDIM;
            int c = 0; float best = row[0];
            for (int j = 1; j < CDIM; ++j) if (row[j] > best) { best = row[j]; c = j; }
            cids[tid] = c;
        }
        __syncthreads();

        // ---- av logits -> gen_enc[:,index,:] ----
        for (int nt = wv; nt < 64; nt += NWAVES) {   // layer1: K=512, N=1024
            v8f c = {};
#pragma unroll 4
            for (int kt = 0; kt < 16; ++kt)
                c = wmma_bf16(load_a_frag(xbuf, 1024, kt, lane),
                              load_b_frag(av1p, 16, nt, kt, lane), c);
            int n = nt * 16 + (lane & 15);
            int mb = (lane >> 4) * 8;
            float bias = av_b1p[n];   // zero-padded; padded cols have zero weights
#pragma unroll
            for (int r = 0; r < 8; ++r) {
                float v = c[r] + bias; v = (v > 0.0f) ? v : 0.0f;
                abuf[(mb + r) * 1024 + n] = (__bf16)v;
            }
        }
        __syncthreads();
        if (wv == 0) {                               // layer2: K=1024, N=16 (C=8)
            v8f c = {};
#pragma unroll 4
            for (int kt = 0; kt < 32; ++kt)
                c = wmma_bf16(load_a_frag(abuf, 1024, kt, lane),
                              load_b_frag(av2p, 32, 0, kt, lane), c);
            int n = lane & 15;
            int mb = (lane >> 4) * 8;
            if (n < CDIM) {
#pragma unroll
                for (int r = 0; r < 8; ++r)
                    out[GEN_ENC_BASE + (size_t)(b0 + mb + r) * 64 + index * 8 + n] =
                        c[r] + av_b2[n];
            }
        }
        __syncthreads();

        // ---- initial GRU of this step ----
        if (index == 0) { gemm_gh(wv, lane, xbuf, whhp, ghws); __threadfence_block(); }
        __syncthreads();
        gru_combine(tid, wihp, bihp, bhhp, ghws, /*use_gh=*/(index == 0),
                    /*hp_mode=*/(index == 0 ? 0 : 1), hvf, acc, xbuf, cids);
        __syncthreads();
        for (int t = tid; t < MTILE * 512; t += NTHREADS) acc[t] = 0.0f;
        __syncthreads();

        // ---- inner loop over previous nodes ----
        for (int v_j = index - 1; v_j >= 0; --v_j) {
            // async DMA: hidden_j (16 rows x 1024B) global -> LDS concat slot [512:1024)
            {
                const unsigned long long gbase =
                    (unsigned long long)(uintptr_t)(nhsb + ((size_t)v_j * MTILE) * 512);
                for (int t = tid; t < MTILE * 64; t += NTHREADS) {
                    int m = t >> 6, k = t & 63;
                    unsigned laddr = lds_xbuf + (unsigned)(m * 2048 + 1024 + k * 16);
                    unsigned long long gaddr = gbase + (unsigned long long)(m * 1024 + k * 16);
                    asm volatile("global_load_async_to_lds_b128 %0, %1, off"
                                 :: "v"(laddr), "v"(gaddr) : "memory");
                }
            }
            if (tid < MTILE) {
                ebuf[tid] = 0.0f;
                depv[tid] = dep[(size_t)(b0 + tid) * 64 + index * 8 + v_j];
            }
            asm volatile("s_wait_asynccnt 0x0" ::: "memory");
            __syncthreads();

            // ae: relu(concat@W1^T+b1) . w2 + b2 (fused reduce), K=1024, N=2016
            float p[8];
#pragma unroll
            for (int r = 0; r < 8; ++r) p[r] = 0.0f;
            for (int nt = wv; nt < 126; nt += NWAVES) {
                if (nt + NWAVES < 126)
                    __builtin_prefetch(ae1p + ((long)(nt + NWAVES) * 32) * 512 + lane * 16, 0, 1);
                v8f c = {};
#pragma unroll 4
                for (int kt = 0; kt < 32; ++kt)
                    c = wmma_bf16(load_a_frag(xbuf, 1024, kt, lane),
                                  load_b_frag(ae1p, 32, nt, kt, lane), c);
                int n = nt * 16 + (lane & 15);
                float b1 = ae_b1p[n];
                float w2 = ae_w2p[n];
#pragma unroll
                for (int r = 0; r < 8; ++r) {
                    float v = c[r] + b1; v = (v > 0.0f) ? v : 0.0f;
                    p[r] += v * w2;
                }
            }
#pragma unroll
            for (int off = 1; off < 16; off <<= 1)
#pragma unroll
                for (int r = 0; r < 8; ++r) p[r] += __shfl_xor(p[r], off, 32);
            if ((lane & 15) == 0) {
                int mb = (lane >> 4) * 8;
#pragma unroll
                for (int r = 0; r < 8; ++r) atomicAdd(&ebuf[mb + r], p[r]);
            }
            __syncthreads();
            if (tid < MTILE)
                out[(size_t)(b0 + tid) * 64 + index * 8 + v_j] = ebuf[tid] + ae_b2[0];

            // h_in prefix sum: acc += dep * gated[v_j]; restage bf16(acc) as A
            for (int t = tid; t < MTILE * 512; t += NTHREADS) {
                int m = t >> 9, n = t & 511;
                float g = (float)gatedb[((size_t)v_j * MTILE + m) * 512 + n];
                float a = acc[t] + depv[m] * g;
                acc[t] = a;
                xbuf[m * 1024 + n] = (__bf16)a;
            }
            __syncthreads();

            // hv = GRU(x_index, h_in)
            gemm_gh(wv, lane, xbuf, whhp, ghws);
            __threadfence_block();
            __syncthreads();
            gru_combine(tid, wihp, bihp, bhhp, ghws, true, /*hp=acc*/2, hvf, acc,
                        xbuf, cids);
            __syncthreads();
        }

        // ---- finalize node: nhs[index]=hv ; gated[index]=sig(gate(hv))*map(hv) ----
        for (int t = tid; t < MTILE * 512; t += NTHREADS) {
            int m = t >> 9, n = t & 511;
            nhsb[((size_t)index * MTILE + m) * 512 + n] = (__bf16)hvf[t];
        }
        for (int nt = wv; nt < 32; nt += NWAVES) {   // gate & map share A fragments
            v8f cg = {}, cm = {};
#pragma unroll 4
            for (int kt = 0; kt < 16; ++kt) {
                v16bf a = load_a_frag(xbuf, 1024, kt, lane);
                cg = wmma_bf16(a, load_b_frag(gatep, 16, nt, kt, lane), cg);
                cm = wmma_bf16(a, load_b_frag(mapp, 16, nt, kt, lane), cm);
            }
            int n = nt * 16 + (lane & 15);
            int mb = (lane >> 4) * 8;
            float gb = gate_bp[n], pb = map_bp[n];
#pragma unroll
            for (int r = 0; r < 8; ++r) {
                float v = sigf(cg[r] + gb) * (cm[r] + pb);
                v = (n < HDIM) ? v : 0.0f;
                gatedb[((size_t)index * MTILE + (mb + r)) * 512 + n] = (__bf16)v;
            }
        }
        __threadfence_block();
        __syncthreads();
    }
}

// ---------------------------------------------------------------------------
extern "C" void kernel_launch(void* const* d_in, const int* in_sizes, int n_in,
                              void* d_out, int out_size, void* d_ws, size_t ws_size,
                              hipStream_t stream) {
    const float* z      = (const float*)d_in[0];
    const float* dep    = (const float*)d_in[1];
    const float* ne     = (const float*)d_in[2];
    const float* lin1_W = (const float*)d_in[3];
    const float* lin1_b = (const float*)d_in[4];
    const float* av_W1  = (const float*)d_in[5];
    const float* av_b1  = (const float*)d_in[6];
    const float* av_W2  = (const float*)d_in[7];
    const float* av_b2  = (const float*)d_in[8];
    const float* ae_W1  = (const float*)d_in[9];
    const float* ae_b1  = (const float*)d_in[10];
    const float* ae_W2  = (const float*)d_in[11];
    const float* ae_b2  = (const float*)d_in[12];
    const float* gate_W = (const float*)d_in[13];
    const float* gate_b = (const float*)d_in[14];
    const float* map_W  = (const float*)d_in[15];
    const float* map_b  = (const float*)d_in[16];
    const float* gru_Wih = (const float*)d_in[17];
    const float* gru_bih = (const float*)d_in[18];
    const float* gru_Whh = (const float*)d_in[19];
    const float* gru_bhh = (const float*)d_in[20];

    __bf16* wsb = (__bf16*)d_ws;
    __bf16* lin1p = wsb + OFF_LIN1;
    __bf16* whhp  = wsb + OFF_WHH;
    __bf16* gatep = wsb + OFF_GATE;
    __bf16* mapp  = wsb + OFF_MAP;
    __bf16* av1p  = wsb + OFF_AV1;
    __bf16* av2p  = wsb + OFF_AV2;
    __bf16* ae1p  = wsb + OFF_AE1;
    __bf16* nhsws = wsb + OFF_NHS;
    __bf16* gatedws = wsb + OFF_GATED;
    float* ghall = (float*)((char*)d_ws + GH_BYTE_OFF);
    float* ptab  = (float*)((char*)d_ws + BIAS_BYTE_OFF);
    float* out = (float*)d_out;

    zero_f32_kernel<<<dim3(128), dim3(256), 0, stream>>>(out, out_size);

    auto pack = [&](const float* W, __bf16* dst, int KsS, int KsP, int NsS, int NsP,
                    int Kp, int Np, int rs) {
        long total = (long)(Kp / 32) * (Np / 16) * 512;
        int blocks = (int)((total + 255) / 256);
        if (blocks > 2048) blocks = 2048;
        pack_b_seg_kernel<<<dim3(blocks), dim3(256), 0, stream>>>(W, dst, KsS, KsP,
                                                                  NsS, NsP, Kp, Np, rs);
    };
    pack(lin1_W, lin1p, LDIM, 64, HDIM, 512, 64, 512, LDIM);
    pack(gru_Whh, whhp, HDIM, 512, HDIM, 512, 512, 1536, HDIM);   // N = 3 segs
    pack(gate_W, gatep, HDIM, 512, HDIM, 512, 512, 512, HDIM);
    pack(map_W, mapp, HDIM, 512, HDIM, 512, 512, 512, HDIM);
    pack(av_W1, av1p, HDIM, 512, H2T, 1024, 512, 1024, HDIM);
    pack(av_W2, av2p, H2T, 1024, CDIM, 16, 1024, 16, H2T);
    pack(ae_W1, ae1p, HDIM, 512, H4T, 2016, 1024, 2016, H2T);     // K = 2 segs

    pad_f32_kernel<<<dim3(2), dim3(256), 0, stream>>>(lin1_b, ptab + POFF_LIN1B, HDIM, 512);
    pad_f32_kernel<<<dim3(4), dim3(256), 0, stream>>>(av_b1, ptab + POFF_AVB1, H2T, 1024);
    pad_f32_kernel<<<dim3(2), dim3(256), 0, stream>>>(gate_b, ptab + POFF_GATEB, HDIM, 512);
    pad_f32_kernel<<<dim3(2), dim3(256), 0, stream>>>(map_b, ptab + POFF_MAPB, HDIM, 512);
    pad_f32_kernel<<<dim3(8), dim3(256), 0, stream>>>(ae_b1, ptab + POFF_AEB1, H4T, 2016);
    pad_f32_kernel<<<dim3(8), dim3(256), 0, stream>>>(ae_W2, ptab + POFF_AEW2, H4T, 2016);
    pad3_kernel<<<dim3(6), dim3(256), 0, stream>>>(gru_bih, ptab + POFF_BIH, 1);
    pad3_kernel<<<dim3(6), dim3(256), 0, stream>>>(gru_bhh, ptab + POFF_BHH, 1);
    pad3_kernel<<<dim3(48), dim3(256), 0, stream>>>(gru_Wih, ptab + POFF_WIH, CDIM);

    decoder_kernel<<<dim3(64), dim3(NTHREADS), 0, stream>>>(
        z, dep, ne, av_b2, ae_b2, ptab, lin1p, whhp, gatep, mapp, av1p, av2p, ae1p,
        nhsws, gatedws, ghall, out);
}
